// MultiHeadAttention_35261681500408
// MI455X (gfx1250) — matrix operand order repaired and checked by reading, not compiled, not run on hardware
//
#include <hip/hip_runtime.h>

// ---------------------------------------------------------------------------
// MI455X (gfx1250, wave32) ProbSparse attention.
// B=4, L=2048, D=256, H=8, DH=32, u=40 (pad to 48 = 3x16 WMMA tiles).
// All GEMM-like steps use v_wmma_f32_16x16x32_bf16 (bf16 in, f32 accum).
// Hot QK step stages gathered K_sample into LDS via async Global->LDS DMA
// (global_load_async_to_lds_b128 / s_wait_asynccnt), shared by 8 waves.
// ---------------------------------------------------------------------------

typedef __attribute__((ext_vector_type(16))) __bf16 v16bf;
typedef __attribute__((ext_vector_type(8)))  __bf16 v8bf;
typedef __attribute__((ext_vector_type(8)))  float  v8f;

#define BD   4
#define LD   2048
#define DD   256
#define HD   8
#define DHD  32
#define BH   32      // B*H
#define UU   40
#define UP   48      // padded u (3 tiles of 16)
#define KSTRIDE 40   // bf16 elements per LDS K_sample row (80B, padded from 64B)

__device__ __forceinline__ v8f wmma_bf16(v16bf a, v16bf b, v8f c) {
  // 8 args: (neg_a, A, neg_b, B, c_mod, C, reuse_a, reuse_b)
  return __builtin_amdgcn_wmma_f32_16x16x32_bf16(false, a, false, b, (short)0, c,
                                                 false, false);
}

// A fragment (16x32 bf16, MxK): lane m = lane%16, hi = lane/16.
// a[e] = A[m][ hi*16 + e ]  -> two contiguous 8-elt (16B) loads.
__device__ __forceinline__ v16bf load_a_bf16(const __bf16* row, int hi) {
  v8bf lo = *reinterpret_cast<const v8bf*>(row + hi * 16);
  v8bf hh = *reinterpret_cast<const v8bf*>(row + hi * 16 + 8);
  v16bf a;
#pragma unroll
  for (int i = 0; i < 8; ++i) { a[i] = lo[i]; a[i + 8] = hh[i]; }
  return a;
}

__device__ __forceinline__ v16bf load_a_f32(const float* row, int hi) {
  v16bf a;
#pragma unroll
  for (int i = 0; i < 8; ++i) {
    a[i]     = (__bf16)row[hi * 16 + i];
    a[i + 8] = (__bf16)row[hi * 16 + 8 + i];
  }
  return a;
}

// B fragment (32x16 bf16, KxN): lane n = lane%16, hi = lane/16.
// b[e] = B[hi*16 + e][n] = (row of B^T)[hi*16 + e] -> one 32B contiguous load.
__device__ __forceinline__ v16bf load_b_bf16(const __bf16* rowT, int hi) {
  return *reinterpret_cast<const v16bf*>(rowT + hi * 16);
}

__device__ __forceinline__ v16bf load_b_f32(const float* rowT, int hi) {
  v16bf b;
#pragma unroll
  for (int e = 0; e < 16; ++e) b[e] = (__bf16)rowT[hi * 16 + e];
  return b;
}

// ---------------------------------------------------------------------------
// K1: Y = X @ W^T + bias, X:[8192,256] f32, W:[256,256] f32 -> Y bf16.
// grid (512, 4, 3); block 128 (4 waves, each a 16-col tile).
// ---------------------------------------------------------------------------
__global__ __launch_bounds__(128)
void proj_kernel(const float* __restrict__ q, const float* __restrict__ k,
                 const float* __restrict__ v,
                 const float* __restrict__ Wq, const float* __restrict__ bq,
                 const float* __restrict__ Wk, const float* __restrict__ bk,
                 const float* __restrict__ Wv, const float* __restrict__ bv,
                 __bf16* __restrict__ qp, __bf16* __restrict__ kp,
                 __bf16* __restrict__ vp) {
  const float* X; const float* W; const float* bias; __bf16* Y;
  if (blockIdx.z == 0)      { X = q; W = Wq; bias = bq; Y = qp; }
  else if (blockIdx.z == 1) { X = k; W = Wk; bias = bk; Y = kp; }
  else                      { X = v; W = Wv; bias = bv; Y = vp; }

  const int lane = threadIdx.x & 31;
  const int wave = threadIdx.x >> 5;
  const int m  = lane & 15;
  const int hi = lane >> 4;
  const int rowBase = blockIdx.x * 16;
  const int colBase = (blockIdx.y * 4 + wave) * 16;

  const float* arow = X + (size_t)(rowBase + m) * DD;
  const float* brow = W + (size_t)(colBase + m) * DD;  // B^T row n = W row n

  v8f c = {};
#pragma unroll
  for (int kk = 0; kk < DD; kk += 32) {
    v16bf a = load_a_f32(arow + kk, hi);
    v16bf b = load_b_f32(brow + kk, hi);
    c = wmma_bf16(a, b, c);
  }
  const float bn = bias[colBase + m];
#pragma unroll
  for (int r = 0; r < 8; ++r) {
    int row = rowBase + r + 8 * hi;
    Y[(size_t)row * DD + colBase + m] = (__bf16)(c[r] + bn);
  }
}

// ---------------------------------------------------------------------------
// K2: M[g][l] = max_u QK[l,u] - sum_u QK[l,u]/L  with K_sample gather.
// One 8-wave block per (head, 128 l-rows). The gathered K_sample slice
// (2048 rows x 32 bf16, padded to 80B rows = 160KB) is staged into LDS once
// per block with async Global->LDS DMA and reused by all 8 waves
// (8x reduction in L2 traffic vs. per-wave streaming).
// grid (16 l-superblocks, 32 heads); block 256; dynamic LDS 160KB.
// ---------------------------------------------------------------------------
__global__ __launch_bounds__(256)
void qk_reduce_kernel(const __bf16* __restrict__ qp,
                      const __bf16* __restrict__ kp,
                      const int* __restrict__ idx_sample,
                      float* __restrict__ Mout) {
  extern __shared__ __bf16 ksmp[];  // [LD][KSTRIDE], dynamic (base offset 0)
  const int tid = threadIdx.x;
  const int g = blockIdx.y;
  const int b = g >> 3, h = g & 7;

  // ---- stage sampled K rows into LDS: 4 lanes x 16B per 64B row ----
  {
    const int chunk = tid & 3;   // which 16B of the row
    const int rbase = tid >> 2;  // row within a 64-row pass
    for (int pass = 0; pass < LD / 64; ++pass) {
      const int r = pass * 64 + rbase;
      const int srcl = idx_sample[r];
      const __bf16* gsrc =
          kp + ((size_t)(b * LD + srcl) * DD + h * DHD) + chunk * 8;
      const unsigned lds_off = (unsigned)(r * (KSTRIDE * 2) + chunk * 16);
      const unsigned long long gaddr = (unsigned long long)(uintptr_t)gsrc;
      asm volatile("global_load_async_to_lds_b128 %0, %1, off"
                   :
                   : "v"(lds_off), "v"(gaddr)
                   : "memory");
    }
    asm volatile("s_wait_asynccnt 0x0" ::: "memory");
  }
  __syncthreads();

  const int lane = tid & 31;
  const int wave = tid >> 5;
  const int n  = lane & 15;
  const int hi = lane >> 4;
  const int l0 = blockIdx.x * 128 + wave * 16;

  // qf = qp viewed flat as [32][2048][32] (faithful to the reference reshape)
  const __bf16* arow = qp + (size_t)(g * LD + l0 + n) * DHD;
  const v16bf a = load_a_bf16(arow, hi);

  float rmax[8], rsum[8];
#pragma unroll
  for (int r = 0; r < 8; ++r) { rmax[r] = -3.402823466e38f; rsum[r] = 0.f; }

  for (int ut = 0; ut < LD / 16; ++ut) {
    const __bf16* brow = &ksmp[(size_t)(ut * 16 + n) * KSTRIDE + hi * 16];
    v16bf bf = *reinterpret_cast<const v16bf*>(brow);
    v8f c = {};
    c = wmma_bf16(a, bf, c);
#pragma unroll
    for (int r = 0; r < 8; ++r) {
      float vv = c[r];
      rmax[r] = fmaxf(rmax[r], vv);
      rsum[r] += vv;
    }
  }
  // reduce across the 16 column-lanes of each half (hi stays separate)
#pragma unroll
  for (int msk = 1; msk < 16; msk <<= 1) {
#pragma unroll
    for (int r = 0; r < 8; ++r) {
      rmax[r] = fmaxf(rmax[r], __shfl_xor(rmax[r], msk, 32));
      rsum[r] += __shfl_xor(rsum[r], msk, 32);
    }
  }
  if (n == 0) {
#pragma unroll
    for (int r = 0; r < 8; ++r) {
      int row = l0 + r + 8 * hi;
      Mout[g * LD + row] = rmax[r] - rsum[r] * (1.0f / (float)LD);
    }
  }
}

// ---------------------------------------------------------------------------
// K3: per-head bottom-40 of M (== top_k(-M)), stable tie-break to lower index.
// grid 32; block 256.
// ---------------------------------------------------------------------------
__global__ __launch_bounds__(256)
void topk_kernel(const float* __restrict__ Mv, int* __restrict__ topIdx) {
  __shared__ float sval[256];
  __shared__ int   sidx[256];
  __shared__ unsigned char taken[LD];
  const int g = blockIdx.x;
  const int tid = threadIdx.x;
  for (int i = tid; i < LD; i += 256) taken[i] = 0;
  __syncthreads();
  const float* Mg = Mv + g * LD;
  for (int it = 0; it < UU; ++it) {
    float best = 3.402823466e38f;
    int bidx = LD;
    for (int i = tid; i < LD; i += 256) {
      if (!taken[i]) {
        float vv = Mg[i];
        if (vv < best || (vv == best && i < bidx)) { best = vv; bidx = i; }
      }
    }
    sval[tid] = best; sidx[tid] = bidx;
    __syncthreads();
    for (int s = 128; s > 0; s >>= 1) {
      if (tid < s) {
        float ov = sval[tid + s]; int oi = sidx[tid + s];
        if (ov < sval[tid] || (ov == sval[tid] && oi < sidx[tid])) {
          sval[tid] = ov; sidx[tid] = oi;
        }
      }
      __syncthreads();
    }
    if (tid == 0) { topIdx[g * UP + it] = sidx[0]; taken[sidx[0]] = 1; }
    __syncthreads();
  }
  if (tid == 0) {  // pad rows 40..47 with a valid index (outputs never stored)
    int last = topIdx[g * UP + UU - 1];
    for (int it = UU; it < UP; ++it) topIdx[g * UP + it] = last;
  }
}

// ---------------------------------------------------------------------------
// K4: scores[g][u'][l] = (Q_red @ kf^T)/sqrt(32); also per-row max.
// grid (3 u'-tiles, 32 heads); block 32.
// ---------------------------------------------------------------------------
__global__ __launch_bounds__(32)
void scores_kernel(const __bf16* __restrict__ qp, const __bf16* __restrict__ kp,
                   const int* __restrict__ topIdx,
                   float* __restrict__ scores, float* __restrict__ rowmax) {
  const int lane = threadIdx.x & 31;
  const int n  = lane & 15;
  const int hi = lane >> 4;
  const int g = blockIdx.y;
  const int b = g >> 3, h = g & 7;
  const int u0 = blockIdx.x * 16;

  const int qrow = topIdx[g * UP + u0 + n];  // A-row m = n
  const v16bf a = load_a_bf16(qp + (size_t)(g * LD + qrow) * DHD, hi);
  const float scale = 0.1767766952966369f;  // 1/sqrt(32)

  float rmax[8];
#pragma unroll
  for (int r = 0; r < 8; ++r) rmax[r] = -3.402823466e38f;

  for (int lt = 0; lt < LD / 16; ++lt) {
    const int l0 = lt * 16;
    const __bf16* brow = kp + ((size_t)(b * LD + l0 + n) * DD + h * DHD);
    v16bf bf = load_b_bf16(brow, hi);
    v8f c = {};
    c = wmma_bf16(a, bf, c);
#pragma unroll
    for (int r = 0; r < 8; ++r) {
      float s = c[r] * scale;
      rmax[r] = fmaxf(rmax[r], s);
      scores[(size_t)(g * UP + u0 + r + 8 * hi) * LD + l0 + n] = s;
    }
  }
#pragma unroll
  for (int msk = 1; msk < 16; msk <<= 1) {
#pragma unroll
    for (int r = 0; r < 8; ++r)
      rmax[r] = fmaxf(rmax[r], __shfl_xor(rmax[r], msk, 32));
  }
  if (n == 0) {
#pragma unroll
    for (int r = 0; r < 8; ++r) rowmax[g * UP + u0 + r + 8 * hi] = rmax[r];
  }
}

// ---------------------------------------------------------------------------
// K5: softmax(scores) @ vf per head; exp() built directly in A-fragment
// layout (lane%16 == A-row), row-sum via shfl, divide at the end.
// grid (3 u'-tiles, 32 heads); block 32.
// ---------------------------------------------------------------------------
__global__ __launch_bounds__(32)
void attn_kernel(const float* __restrict__ scores,
                 const float* __restrict__ rowmax,
                 const __bf16* __restrict__ vp, __bf16* __restrict__ attn) {
  const int lane = threadIdx.x & 31;
  const int m  = lane & 15;
  const int hi = lane >> 4;
  const int g = blockIdx.y;
  const int b = g >> 3, h = g & 7;
  const int u0 = blockIdx.x * 16;

  const float* srow = scores + (size_t)(g * UP + u0 + m) * LD;
  const float rm = rowmax[g * UP + u0 + m];

  v8f acc0 = {}, acc1 = {};
  float psum = 0.f;

  for (int ch = 0; ch < LD / 32; ++ch) {
    const int l0 = ch * 32;
    v16bf a;
#pragma unroll
    for (int e = 0; e < 16; ++e) {
      int K = hi * 16 + e;                          // A-fragment K mapping
      float p = __expf(srow[l0 + K] - rm);
      psum += p;
      a[e] = (__bf16)p;
    }
    v16bf b0, b1;
#pragma unroll
    for (int e = 0; e < 16; ++e) {
      int K = hi * 16 + e;                          // B-fragment K mapping
      const __bf16* vr = vp + ((size_t)(b * LD + l0 + K) * DD + h * DHD);
      b0[e] = vr[m];
      b1[e] = vr[16 + m];
    }
    acc0 = wmma_bf16(a, b0, acc0);
    acc1 = wmma_bf16(a, b1, acc1);
  }
  // lanes m and m+16 jointly cover all K for row m
  psum += __shfl_xor(psum, 16, 32);
#pragma unroll
  for (int r = 0; r < 8; ++r) {
    const int urow = u0 + r + 8 * hi;
    const float rs = __shfl(psum, r + 8 * hi, 32);
    if (urow < UU) {
      size_t base = (size_t)(b * UU + urow) * DD + h * DHD;
      attn[base + m]      = (__bf16)(acc0[r] / rs);
      attn[base + 16 + m] = (__bf16)(acc1[r] / rs);
    }
  }
}

// ---------------------------------------------------------------------------
// K6: out = attn @ Wc^T + bc, [160,256] @ [256,256]^T, f32 output.
// grid (10, 16); block 32.
// ---------------------------------------------------------------------------
__global__ __launch_bounds__(32)
void final_kernel(const __bf16* __restrict__ attn, const float* __restrict__ Wc,
                  const float* __restrict__ bc, float* __restrict__ out) {
  const int lane = threadIdx.x & 31;
  const int m  = lane & 15;
  const int hi = lane >> 4;
  const int rowBase = blockIdx.x * 16;
  const int colBase = blockIdx.y * 16;

  const __bf16* arow = attn + (size_t)(rowBase + m) * DD;
  const float*  brow = Wc + (size_t)(colBase + m) * DD;

  v8f c = {};
#pragma unroll
  for (int kk = 0; kk < DD; kk += 32) {
    v16bf a = load_a_bf16(arow + kk, hi);
    v16bf b = load_b_f32(brow + kk, hi);
    c = wmma_bf16(a, b, c);
  }
  const float bn = bc[colBase + m];
#pragma unroll
  for (int r = 0; r < 8; ++r) {
    int row = rowBase + r + 8 * hi;
    out[(size_t)row * DD + colBase + m] = c[r] + bn;
  }
}

// ---------------------------------------------------------------------------
extern "C" void kernel_launch(void* const* d_in, const int* in_sizes, int n_in,
                              void* d_out, int out_size, void* d_ws,
                              size_t ws_size, hipStream_t stream) {
  (void)in_sizes; (void)n_in; (void)out_size; (void)ws_size;
  const float* q  = (const float*)d_in[0];
  const float* k  = (const float*)d_in[1];
  const float* v  = (const float*)d_in[2];
  const float* Wq = (const float*)d_in[3];
  const float* bq = (const float*)d_in[4];
  const float* Wk = (const float*)d_in[5];
  const float* bk = (const float*)d_in[6];
  const float* Wv = (const float*)d_in[7];
  const float* bv = (const float*)d_in[8];
  const float* Wc = (const float*)d_in[9];
  const float* bc = (const float*)d_in[10];
  const int* idxs = (const int*)d_in[11];
  float* out = (float*)d_out;

  char* ws = (char*)d_ws;
  size_t off = 0;
  auto take = [&](size_t bytes) {
    char* p = ws + off;
    off += (bytes + 255) & ~(size_t)255;
    return p;
  };
  __bf16* qp      = (__bf16*)take((size_t)BD * LD * DD * 2);   // 4 MB
  __bf16* kp      = (__bf16*)take((size_t)BD * LD * DD * 2);   // 4 MB
  __bf16* vp      = (__bf16*)take((size_t)BD * LD * DD * 2);   // 4 MB
  float*  scoresB = (float*)take((size_t)BH * UP * LD * 4);    // 12 MB
  float*  Mbuf    = (float*)take((size_t)BH * LD * 4);         // 256 KB
  float*  rowmaxB = (float*)take((size_t)BH * UP * 4);
  int*    topIdx  = (int*)take((size_t)BH * UP * 4);
  __bf16* attnB   = (__bf16*)take((size_t)BD * UU * DD * 2);

  proj_kernel<<<dim3(512, 4, 3), 128, 0, stream>>>(q, k, v, Wq, bq, Wk, bk, Wv,
                                                   bv, qp, kp, vp);
  const size_t kLds = (size_t)LD * KSTRIDE * 2;  // 160 KB dynamic LDS
  qk_reduce_kernel<<<dim3(LD / 128, BH), 256, kLds, stream>>>(qp, kp, idxs,
                                                              Mbuf);
  topk_kernel<<<BH, 256, 0, stream>>>(Mbuf, topIdx);
  scores_kernel<<<dim3(UP / 16, BH), 32, 0, stream>>>(qp, kp, topIdx, scoresB,
                                                      rowmaxB);
  attn_kernel<<<dim3(UP / 16, BH), 32, 0, stream>>>(scoresB, rowmaxB, vp, attnB);
  final_kernel<<<dim3((BD * UU) / 16, DD / 16), 32, 0, stream>>>(attnB, Wc, bc,
                                                                 out);
}
